// CompositeReservoirCell_10539849744900
// MI455X (gfx1250) — matrix operand order repaired and checked by reading, not compiled, not run on hardware
//
#include <hip/hip_runtime.h>

typedef __attribute__((ext_vector_type(16))) _Float16 v16h;
typedef __attribute__((ext_vector_type(8)))  float    v8f;

#define B_DIM 16
#define I_DIM 1024
#define H_DIM 2048

// ---------------------------------------------------------------------------
// Kernel 1: input_contrib = x_t (16x1024) @ W_ih_w^T (1024x2048) + bias
// One block (4 waves) per 16-column N-tile; each wave owns a K-quarter (256),
// 8 x v_wmma_f32_16x16x32_f16 per wave, LDS reduction across the 4 waves.
// C[m,n] = sum_k x[m,k] * W[n,k]  ->  A = x (16xK), B[k,n] = W[n,k].
// ---------------------------------------------------------------------------
__device__ __forceinline__ void cvt8(v16h& d, int base, float4 a, float4 b) {
    d[base + 0] = (_Float16)a.x; d[base + 1] = (_Float16)a.y;
    d[base + 2] = (_Float16)a.z; d[base + 3] = (_Float16)a.w;
    d[base + 4] = (_Float16)b.x; d[base + 5] = (_Float16)b.y;
    d[base + 6] = (_Float16)b.z; d[base + 7] = (_Float16)b.w;
}

__global__ __launch_bounds__(128) void input_gemm_wmma(
    const float* __restrict__ x,     // (16,1024)
    const float* __restrict__ W,     // (2048,1024)
    const float* __restrict__ bias,  // (2048,)
    float* __restrict__ out)         // (16,2048)
{
    __shared__ float partial[4][32][8];

    const int lane = threadIdx.x & 31;
    const int wv   = threadIdx.x >> 5;     // K-split index 0..3
    const int n0   = blockIdx.x << 4;      // N-tile base
    const int m    = lane & 15;            // A row / C column-within-tile
    const int hi   = lane >> 4;            // lane half: 0 or 1
    const int n    = n0 + m;               // B column (row of W)

    // ISA layouts (16-bit, wave32):
    //  A 16x32: lanes 0-15 hold K {0..7, 16..23}, lanes 16-31 hold K {8..15, 24..31}
    //  B 32x16: lanes 0-15 hold K {0..7, 16..23} of col N=lane, lanes 16-31 hold K {8..15, 24..31}
    const int a_koff = hi << 3;            // 0 or 8
    const int b_koff = hi << 3;            // 0 or 8

    v8f c = {};
    #pragma unroll
    for (int kk = 0; kk < 8; ++kk) {
        const int k0 = (wv << 8) + (kk << 5);   // this wave's K chunk base

        const float* arow = x + m * I_DIM + k0 + a_koff;
        float4 a0 = *reinterpret_cast<const float4*>(arow + 0);
        float4 a1 = *reinterpret_cast<const float4*>(arow + 4);
        float4 a2 = *reinterpret_cast<const float4*>(arow + 16);
        float4 a3 = *reinterpret_cast<const float4*>(arow + 20);

        const float* brow = W + n * I_DIM + k0 + b_koff;
        float4 b0 = *reinterpret_cast<const float4*>(brow + 0);
        float4 b1 = *reinterpret_cast<const float4*>(brow + 4);
        float4 b2 = *reinterpret_cast<const float4*>(brow + 16);
        float4 b3 = *reinterpret_cast<const float4*>(brow + 20);

        v16h av, bv;
        cvt8(av, 0, a0, a1);  cvt8(av, 8, a2, a3);
        cvt8(bv, 0, b0, b1);  cvt8(bv, 8, b2, b3);

        c = __builtin_amdgcn_wmma_f32_16x16x32_f16(
                /*neg_a=*/false, av, /*neg_b=*/false, bv,
                /*c_mod=*/(short)0, c, /*reuse_a=*/false, /*reuse_b=*/false);
    }

    #pragma unroll
    for (int r = 0; r < 8; ++r) partial[wv][lane][r] = c[r];
    __syncthreads();

    if (wv == 0) {
        const float bval = bias[n];
        #pragma unroll
        for (int r = 0; r < 8; ++r) {
            float v = partial[0][lane][r] + partial[1][lane][r]
                    + partial[2][lane][r] + partial[3][lane][r];
            // C layout: VGPR r -> M = r (lanes 0-15) / M = 8+r (lanes 16-31)
            const int mo = r + (hi << 3);
            out[mo * H_DIM + n] = v + bval;
        }
    }
}

// ---------------------------------------------------------------------------
// Kernel 2: streaming Gumbel-softmax weighted sum + tanh, one wave per (b,o).
//   z[h]   = W_hh[o,h]*mask[o,h]/tau + (-log(-log(u[b,o,h]+eps)+eps))
//   out    = tanh(input_contrib[b,o] + sum(e^z * h_prev[b,h]) / sum(e^z))
// Single pass over the 268 MB noise tensor (HBM-bound); unstabilized sum is
// safe because the eps guards bound z (exp(z) <= ~1e10 << FLT_MAX).
// ---------------------------------------------------------------------------
__global__ __launch_bounds__(256) void gumbel_softmax_row(
    const float* __restrict__ h_prev,   // (16,2048)
    const float* __restrict__ W_hh,     // (2048,2048)
    const float* __restrict__ hh_mask,  // (2048,2048)
    const float* __restrict__ temp,     // (1,)
    const float* __restrict__ u,        // (16,2048,2048)
    float* __restrict__ out)            // (16,2048): in = input_contrib, out = h_next
{
    const int lane = threadIdx.x & 31;
    const int wv   = threadIdx.x >> 5;
    const int row  = (blockIdx.x << 3) + wv;    // 0..32767  (= b*2048 + o)
    const int b    = row >> 11;
    const int o    = row & (H_DIM - 1);

    const float inv_tau = 1.0f / fmaxf(temp[0], 1e-3f);

    const float4* u4 = reinterpret_cast<const float4*>(u       + ((size_t)row << 11));
    const float4* w4 = reinterpret_cast<const float4*>(W_hh    + ((size_t)o   << 11));
    const float4* m4 = reinterpret_cast<const float4*>(hh_mask + ((size_t)o   << 11));
    const float4* h4 = reinterpret_cast<const float4*>(h_prev  + ((size_t)b   << 11));

    float s  = 0.0f;   // sum of exp(z)
    float ws = 0.0f;   // sum of exp(z) * h_prev

    #pragma unroll 4
    for (int i = 0; i < 16; ++i) {
        const int idx = (i << 5) + lane;        // float4 index, coalesced
        const float4 uu = u4[idx];
        const float4 ww = w4[idx];
        const float4 mm = m4[idx];
        const float4 hh = h4[idx];

        float g, e;
        g = -__logf(-__logf(uu.x + 1e-10f) + 1e-10f);
        e = __expf(fmaf(ww.x * mm.x, inv_tau, g));  s += e;  ws = fmaf(e, hh.x, ws);
        g = -__logf(-__logf(uu.y + 1e-10f) + 1e-10f);
        e = __expf(fmaf(ww.y * mm.y, inv_tau, g));  s += e;  ws = fmaf(e, hh.y, ws);
        g = -__logf(-__logf(uu.z + 1e-10f) + 1e-10f);
        e = __expf(fmaf(ww.z * mm.z, inv_tau, g));  s += e;  ws = fmaf(e, hh.z, ws);
        g = -__logf(-__logf(uu.w + 1e-10f) + 1e-10f);
        e = __expf(fmaf(ww.w * mm.w, inv_tau, g));  s += e;  ws = fmaf(e, hh.w, ws);
    }

    // wave32 butterfly reduction
    #pragma unroll
    for (int off = 16; off > 0; off >>= 1) {
        s  += __shfl_xor(s,  off, 32);
        ws += __shfl_xor(ws, off, 32);
    }

    if (lane == 0) {
        out[row] = tanhf(out[row] + ws / s);
    }
}

// ---------------------------------------------------------------------------
extern "C" void kernel_launch(void* const* d_in, const int* in_sizes, int n_in,
                              void* d_out, int out_size, void* d_ws, size_t ws_size,
                              hipStream_t stream) {
    (void)in_sizes; (void)n_in; (void)out_size; (void)d_ws; (void)ws_size;

    const float* x_t         = (const float*)d_in[0];  // (16,1024)
    const float* h_prev      = (const float*)d_in[1];  // (16,2048)
    const float* W_ih_w      = (const float*)d_in[2];  // (2048,1024)
    const float* W_ih_b      = (const float*)d_in[3];  // (2048,)
    const float* W_hh        = (const float*)d_in[4];  // (2048,2048)
    const float* hh_mask     = (const float*)d_in[5];  // (2048,2048)
    const float* temperature = (const float*)d_in[6];  // (1,)
    const float* gumbel      = (const float*)d_in[7];  // (16,2048,2048)
    float* out               = (float*)d_out;          // (16,2048)

    // Stage 1: input_contrib -> d_out (128 N-tiles, 4 K-split waves each)
    input_gemm_wmma<<<dim3(H_DIM / 16), dim3(128), 0, stream>>>(
        x_t, W_ih_w, W_ih_b, out);

    // Stage 2: streaming gumbel-softmax row kernel (one wave per (b,o) row)
    gumbel_softmax_row<<<dim3((B_DIM * H_DIM) / 8), dim3(256), 0, stream>>>(
        h_prev, W_hh, hh_mask, temperature, gumbel, out);
}
